// COVHOPF_41171556499997
// MI455X (gfx1250) — matrix operand order repaired and checked by reading, not compiled, not run on hardware
//
#include <hip/hip_runtime.h>

// Coupled Hopf oscillators, persistent single-WGP kernel for MI455X (gfx1250).
// N=500 (padded to 512), T=20000 Euler-Maruyama steps.
// Matvec sc@x / sc@y via v_wmma_f32_16x16x32_bf16; SC (bf16, 512KB) held fully
// on-chip: 8 K-chunks/wave in registers (64 VGPRs), 8 K-chunks in LDS (256KB).

#define NN   500
#define NP   512
#define TT   20000
#define DTc  0.05f
#define SQDTc 0.22360679774997896f   // sqrt(0.05)

#define KREG 8    // K-chunks 0..7  resident in registers (64 VGPRs)
#define KLDS 8    // K-chunks 8..15 resident in LDS (pre-swizzled, 8KB/wave)

typedef __attribute__((ext_vector_type(16))) __bf16 v16bf;
typedef __attribute__((ext_vector_type(8)))  float  v8f;

union BFrag { v16bf v; unsigned short s[16]; };

__device__ __forceinline__ unsigned short f2bf(float f) {
  union { float f; unsigned int u; } c; c.f = f;
  unsigned int u = c.u;
  unsigned int r = u + 0x7FFFu + ((u >> 16) & 1u);   // round to nearest even
  return (unsigned short)(r >> 16);
}

__launch_bounds__(1024, 1)
__global__ void COVHOPF_hopf_persistent(const float* __restrict__ sc,
                                        const float* __restrict__ x0,
                                        const float* __restrict__ y0,
                                        const float* __restrict__ aArr,
                                        const float* __restrict__ omArr,
                                        const float* __restrict__ noise,
                                        const float* __restrict__ gArr,
                                        float* __restrict__ out)
{
  // --- LDS: 256KB swizzled A fragments + state vectors + coupling results ---
  __shared__ __align__(32) char           ldsA[32 * KLDS * 1024]; // 1KB per (wave,kc)
  __shared__ __align__(32) unsigned short xb[NP];                 // bf16 x state
  __shared__ __align__(32) unsigned short yb[NP];                 // bf16 y state
  __shared__ __align__(32) float          cxA[NP];                // sc@x
  __shared__ __align__(32) float          cyA[NP];                // sc@y

  const int tid  = threadIdx.x;
  const int wave = tid >> 5;          // 0..31 : M-tile owned by this wave
  const int lane = tid & 31;
  const int mcol = lane & 15;         // B/C column index (N)
  const int h    = lane >> 4;         // lane-half per WMMA layout
  const int mrow = wave * 16 + mcol;  // SC row this lane loads for A (M = lane&15)

  // --- persistent per-row fp32 state (thread tid owns row tid, tid<500) ---
  float x = 0.f, y = 0.f, av = 0.f, om = 0.f, deg = 0.f;
  const float g = gArr[0];
  if (tid < NP) {
    if (tid < NN) {
      x  = x0[tid];  y  = y0[tid];
      av = aArr[tid]; om = omArr[tid];
      const float* rowp = sc + (long)tid * NN;
      float s = 0.f;
      for (int j = 0; j < NN; ++j) s += rowp[j];
      deg = s;                              // in-strength (≈1, row-normalized)
    }
    xb[tid] = f2bf(x);                      // rows >=500 publish zeros forever
    yb[tid] = f2bf(y);
  }

  // --- build A fragments (SC rows [16w,16w+16), bf16, zero-padded) ---
  // 16-bit A 16x32 layout: lane(M=lane&15, h=lane>>4);
  //   elems 0..7  -> K = 8h + i ; elems 8..15 -> K = 16 + 8h + (i-8)
  v16bf Areg[KREG];
  {
    const bool rowOK = (mrow < NN);
    const float* rowp = sc + (long)mrow * NN;
    #pragma unroll
    for (int kc = 0; kc < 16; ++kc) {
      BFrag f;
      #pragma unroll
      for (int i = 0; i < 8; ++i) {
        const int ca = kc * 32 + h * 8 + i;
        const int cb = kc * 32 + 16 + h * 8 + i;
        f.s[i]     = f2bf((rowOK && ca < NN) ? rowp[ca] : 0.f);
        f.s[8 + i] = f2bf((rowOK && cb < NN) ? rowp[cb] : 0.f);
      }
      if (kc < KREG) {
        Areg[kc] = f.v;
      } else {
        v16bf* dst = (v16bf*)(ldsA + ((wave * KLDS + (kc - KREG)) << 10) + lane * 32);
        *dst = f.v;                          // pre-swizzled: 32B/lane, 1KB/fragment
      }
    }
  }
  __syncthreads();

  // B source: column 1 reads y, all other columns read x (cols 2..15 unused in C)
  const unsigned short* bsrc = (mcol == 1) ? yb : xb;
  const char* ldsAw = ldsA + ((wave * KLDS) << 10) + lane * 32;

  for (int t = 0; t < TT; ++t) {
    // prefetch NEXT step's noise line: global_prefetch_b8, no VGPR, no wait.
    if (tid < NN) {
      const int tp = (t + 1 < TT) ? (t + 1) : t;
      __builtin_prefetch(noise + (long)tp * (2 * NN) + 2 * tid, 0, 1);
    }

    // --- matvec: C[16x16] = A[16x512] * B[512x16], two interleaved accumulators ---
    v8f c0 = {}; v8f c1 = {};
    #pragma unroll
    for (int kc = 0; kc < 16; ++kc) {
      // B(32x16) layout: lane = column; elems 0..7 -> K = 8h + i ; elems 8..15 -> K = 16 + 8h + (i-8)
      typedef __attribute__((ext_vector_type(8))) __bf16 v8bf_;
      union { v16bf v; v8bf_ p[2]; } bu;
      bu.p[0] = *(const v8bf_*)(bsrc + kc * 32 + h * 8);
      bu.p[1] = *(const v8bf_*)(bsrc + kc * 32 + 16 + h * 8);
      v16bf b = bu.v;
      v16bf aF;
      if (kc < KREG) aF = Areg[kc];
      else aF = *(const v16bf*)(ldsAw + ((kc - KREG) << 10));
      if (kc & 1)
        c1 = __builtin_amdgcn_wmma_f32_16x16x32_bf16(false, aF, false, b,
                                                     (short)0, c1, false, false);
      else
        c0 = __builtin_amdgcn_wmma_f32_16x16x32_bf16(false, aF, false, b,
                                                     (short)0, c0, false, false);
    }
    v8f cs = c0 + c1;

    // C layout: VGPR r -> M = 8h + r, lanes N = mcol. Columns 0/1 hold sc@x / sc@y.
    if (mcol < 2) {
      float* arr = mcol ? cyA : cxA;
      *(v8f*)&arr[wave * 16 + h * 8] = cs;   // 8 consecutive rows, one 32B LDS store
    }
    __syncthreads();

    // --- Hopf dynamics (fp32, thread tid owns row tid) ---
    if (tid < NN) {
      const float* np = noise + (long)t * (2 * NN) + 2 * tid;
      const float nz0 = __builtin_nontemporal_load(np);      // L0 hit (prefetched)
      const float nz1 = __builtin_nontemporal_load(np + 1);
      const float cx = cxA[tid] - deg * x;
      const float cy = cyA[tid] - deg * y;
      const float r2 = x * x + y * y;
      const float dx = (av - r2) * x - om * y + g * cx;
      const float dy = (av - r2) * y + om * x + g * cy;
      x = x + DTc * dx + SQDTc * nz0;
      y = y + DTc * dy + SQDTc * nz1;
      __builtin_nontemporal_store(x, out + (long)t * NN + tid);
      xb[tid] = f2bf(x);
      yb[tid] = f2bf(y);
    }
    __syncthreads();
  }
}

extern "C" void kernel_launch(void* const* d_in, const int* in_sizes, int n_in,
                              void* d_out, int out_size, void* d_ws, size_t ws_size,
                              hipStream_t stream) {
  (void)in_sizes; (void)n_in; (void)out_size; (void)d_ws; (void)ws_size;
  const float* sc    = (const float*)d_in[0];
  const float* x0    = (const float*)d_in[1];
  const float* y0    = (const float*)d_in[2];
  const float* aArr  = (const float*)d_in[3];
  const float* omArr = (const float*)d_in[4];
  const float* noise = (const float*)d_in[5];
  const float* gArr  = (const float*)d_in[6];
  float* out = (float*)d_out;

  // Single persistent workgroup: 1024 threads = 32 wave32s on one WGP.
  COVHOPF_hopf_persistent<<<1, 1024, 0, stream>>>(sc, x0, y0, aArr, omArr,
                                                  noise, gArr, out);
}
